// LSTMModel_84902913507804
// MI455X (gfx1250) — matrix operand order repaired and checked by reading, not compiled, not run on hardware
//
#include <hip/hip_runtime.h>
#include <hip/hip_bf16.h>
#include <stdint.h>

// ---------------------------------------------------------------------------
// Problem constants (from reference)
// ---------------------------------------------------------------------------
#define BB   512
#define TT   256
#define INP  64
#define HH   256
#define GG   1024   // 4*HH

typedef __attribute__((ext_vector_type(16))) _Float16     v16h;
typedef __attribute__((ext_vector_type(8)))  _Float16     v8h;
typedef __attribute__((ext_vector_type(8)))  float        v8f;
typedef __attribute__((ext_vector_type(4)))  unsigned int v4u;

// global (addrspace 1) pointer to a 16-byte chunk (builtin vector type, so
// assignment from an addrspace(1) lvalue is a plain value load)
typedef __attribute__((address_space(1))) const v4u GU4;

union AV { v16h v; v8h h[2]; };
union BV { v16h v; v4u u[2]; };
union PK { _Float16 h[2]; uint32_t u; };

__device__ __forceinline__ float sigmoidf(float x) {
    return 1.0f / (1.0f + __expf(-x));
}
// tanh(x) = 2*sigmoid(2x) - 1  (stable: saturates to +/-1)
__device__ __forceinline__ float tanh_f(float x) {
    return __builtin_fmaf(2.0f, sigmoidf(2.0f * x), -1.0f);
}

// ---------------------------------------------------------------------------
// f32 -> f16 flat convert
// ---------------------------------------------------------------------------
__global__ void k_f32_to_f16(const float* __restrict__ in, _Float16* __restrict__ out, int n) {
    int i = blockIdx.x * blockDim.x + threadIdx.x;
    if (i < n) out[i] = (_Float16)in[i];
}

// ---------------------------------------------------------------------------
// bias combine: out[g] = b_ih[g] + b_hh[g]
// ---------------------------------------------------------------------------
__global__ void k_bias_combine(const float* __restrict__ bi, const float* __restrict__ bh,
                               float* __restrict__ out) {
    int i = blockIdx.x * blockDim.x + threadIdx.x;
    if (i < GG) out[i] = bi[i] + bh[i];
}

// ---------------------------------------------------------------------------
// Pack weight [G=1024, K] (row-major f32) into B-fragment layout for
// v_wmma_f32_16x16x32_f16.  Fragment (ntile, ktile): per lane 8 dwords
// (16 halves).  lane 0-15: N = lane,  K = kt*32 + 2v,2v+1
// lane 16-31:             N = lane-16, K = kt*32 + 16 + 2v,2v+1
// out layout: [ntile][ktile][lane][8] dwords
// ---------------------------------------------------------------------------
__global__ void k_pack_B(const float* __restrict__ w, uint32_t* __restrict__ out, int K) {
    int idx = blockIdx.x * blockDim.x + threadIdx.x;
    int KT = K >> 5;
    int total = (GG / 16) * KT * 32 * 8;
    if (idx >= total) return;
    int v    = idx & 7;
    int lane = (idx >> 3) & 31;
    int frag = idx >> 8;
    int kt    = frag % KT;
    int ntile = frag / KT;
    int n  = ntile * 16 + (lane & 15);
    int kb = kt * 32 + ((lane >> 4) * 16) + v * 2;
    PK p;
    p.h[0] = (_Float16)w[(size_t)n * K + kb + 0];
    p.h[1] = (_Float16)w[(size_t)n * K + kb + 1];
    out[idx] = p.u;
}

// ---------------------------------------------------------------------------
// Persistent per-layer LSTM kernel.
//   grid = B/16 workgroups (each owns 16 batch rows, all T steps)
//   block = 256 threads = 8 waves; wave w owns cell columns [32w, 32w+32)
//   gates = bias + x_t @ w_ih^T + h_{t-1} @ w_hh^T   via WMMA f16->f32
//   h kept in LDS (double buffered), c in registers.
// ---------------------------------------------------------------------------
template <int DIN>
__global__ __launch_bounds__(256)
void k_lstm_layer(const _Float16* __restrict__ xin,      // [B,T,DIN] f16
                  const uint32_t* __restrict__ wih_pk,   // packed frags, K=DIN
                  const uint32_t* __restrict__ whh_pk,   // packed frags, K=HH
                  const float*    __restrict__ bias,     // [G] combined
                  _Float16*       __restrict__ hs)       // [B,T,H] f16 out
{
    constexpr int KX = DIN / 32;      // x  K-tiles
    constexpr int KH = HH  / 32;      // h  K-tiles (8)
    constexpr int SX = DIN + 8;       // LDS row stride (halves), bank-rotating
    constexpr int SH = HH  + 8;

    __shared__ __align__(16) _Float16 smem[16 * SX + 2 * 16 * SH];
    _Float16* xb    = smem;
    _Float16* hbuf0 = smem + 16 * SX;
    _Float16* hbuf1 = smem + 16 * SX + 16 * SH;

    const int tid  = threadIdx.x;
    const int lane = tid & 31;
    const int wv   = tid >> 5;            // wave id 0..7
    const int b0   = blockIdx.x * 16;     // batch row base
    const int m    = lane & 15;           // N index within tile / A row
    const int hi   = lane >> 4;           // half-wave select
    const int c0   = wv * 32;             // this wave's first cell column

    // zero LDS (h_{-1} = 0)
    for (int i = tid; i < 16 * SX + 2 * 16 * SH; i += 256) smem[i] = (_Float16)0;
    __syncthreads();

    // cell state c for this wave's 2 column tiles, C/D register layout
    v8f cst[2];
    #pragma unroll
    for (int tt = 0; tt < 2; ++tt)
        #pragma unroll
        for (int r = 0; r < 8; ++r) cst[tt][r] = 0.0f;

    // bias per accumulator tile (depends only on column)
    float bval[8];
    #pragma unroll
    for (int q = 0; q < 4; ++q)
        #pragma unroll
        for (int tt = 0; tt < 2; ++tt)
            bval[q * 2 + tt] = bias[q * HH + c0 + tt * 16 + m];

    for (int t = 0; t < TT; ++t) {
        const _Float16* hrd = (t & 1) ? hbuf1 : hbuf0;
        _Float16*       hwr = (t & 1) ? hbuf0 : hbuf1;

        // Launder the packed-weight base pointers each timestep so LICM cannot
        // hoist the fragment loads out of the t-loop (which caused scratch
        // spills).  Keep them in addrspace(1) so the backend emits
        // global_load_b128 (saddr + 32-bit voffset), not flat_load.
        GU4* wih_t = (GU4*)wih_pk;
        GU4* whh_t = (GU4*)whh_pk;
        asm volatile("" : "+s"(wih_t), "+s"(whh_t));

        // ---- stage x_t tile [16][DIN] into LDS (dword granularity) ----
        {
            const uint32_t* xs = (const uint32_t*)xin;
            #pragma unroll 4
            for (int i = tid; i < 16 * (DIN / 2); i += 256) {
                int row = i / (DIN / 2);
                int kk  = i % (DIN / 2);
                uint32_t val = xs[((size_t)(b0 + row) * TT + t) * (DIN / 2) + kk];
                *(uint32_t*)(xb + row * SX + 2 * kk) = val;
            }
        }
        __syncthreads();

        // ---- init accumulators with bias ----
        v8f acc[8];
        #pragma unroll
        for (int a8 = 0; a8 < 8; ++a8)
            #pragma unroll
            for (int r = 0; r < 8; ++r) acc[a8][r] = bval[a8];

        // ---- K loop over x part: gates += x_t @ w_ih^T ----
        #pragma unroll
        for (int kt = 0; kt < KX; ++kt) {
            AV a;
            a.h[0] = *(const v8h*)(xb + m * SX + kt * 32 + hi * 16);
            a.h[1] = *(const v8h*)(xb + m * SX + kt * 32 + hi * 16 + 8);
            #pragma unroll
            for (int q = 0; q < 4; ++q)
                #pragma unroll
                for (int tt = 0; tt < 2; ++tt) {
                    const int ntile = q * 16 + wv * 2 + tt;
                    const int idx = ((ntile * KX + kt) * 32 + lane) * 2;  // 16B units
                    BV b;
                    b.u[0] = wih_t[idx + 0];
                    b.u[1] = wih_t[idx + 1];
                    acc[q * 2 + tt] = __builtin_amdgcn_wmma_f32_16x16x32_f16(
                        false, a.v, false, b.v, (short)0, acc[q * 2 + tt], false, false);
                }
        }

        // ---- K loop over h part: gates += h_{t-1} @ w_hh^T ----
        #pragma unroll
        for (int kt = 0; kt < KH; ++kt) {
            AV a;
            a.h[0] = *(const v8h*)(hrd + m * SH + kt * 32 + hi * 16);
            a.h[1] = *(const v8h*)(hrd + m * SH + kt * 32 + hi * 16 + 8);
            #pragma unroll
            for (int q = 0; q < 4; ++q)
                #pragma unroll
                for (int tt = 0; tt < 2; ++tt) {
                    const int ntile = q * 16 + wv * 2 + tt;
                    const int idx = ((ntile * KH + kt) * 32 + lane) * 2;  // 16B units
                    BV b;
                    b.u[0] = whh_t[idx + 0];
                    b.u[1] = whh_t[idx + 1];
                    acc[q * 2 + tt] = __builtin_amdgcn_wmma_f32_16x16x32_f16(
                        false, a.v, false, b.v, (short)0, acc[q * 2 + tt], false, false);
                }
        }

        // ---- LSTM cell update for this wave's 32 cells x 16 rows ----
        #pragma unroll
        for (int tt = 0; tt < 2; ++tt) {
            const int col = c0 + tt * 16 + m;
            #pragma unroll
            for (int r = 0; r < 8; ++r) {
                float ig = sigmoidf(acc[0 * 2 + tt][r]);
                float fg = sigmoidf(acc[1 * 2 + tt][r]);
                float gg = tanh_f (acc[2 * 2 + tt][r]);
                float og = sigmoidf(acc[3 * 2 + tt][r]);
                float c  = fg * cst[tt][r] + ig * gg;
                cst[tt][r] = c;
                float h  = og * tanh_f(c);
                const int row = hi * 8 + r;                  // C/D layout: lanes16-31 -> M=8..15
                _Float16 h16 = (_Float16)h;
                hwr[row * SH + col] = h16;                   // ds_store for next step
                hs[((size_t)(b0 + row) * TT + t) * HH + col] = h16;  // layer output
            }
        }
        __syncthreads();
    }
}

// ---------------------------------------------------------------------------
// FC head: out[b] = fc2_b + sum_j relu(fc1_b[j] + h_last[b,:]·fc1_w[j,:]) * fc2_w[j]
// ---------------------------------------------------------------------------
__global__ void k_fc_head(const _Float16* __restrict__ hs,   // [B,T,H] f16
                          const float* __restrict__ w1, const float* __restrict__ b1,
                          const float* __restrict__ w2, const float* __restrict__ b2,
                          float* __restrict__ out)
{
    int b = blockIdx.x * blockDim.x + threadIdx.x;
    if (b >= BB) return;
    const _Float16* hv = hs + ((size_t)b * TT + (TT - 1)) * HH;
    float accum = b2[0];
    for (int j = 0; j < 25; ++j) {
        float s = b1[j];
        #pragma unroll 8
        for (int k = 0; k < HH; ++k) s += (float)hv[k] * w1[j * HH + k];
        accum += fmaxf(s, 0.0f) * w2[j];
    }
    out[b] = accum;
}

// ---------------------------------------------------------------------------
// Launcher
// ---------------------------------------------------------------------------
extern "C" void kernel_launch(void* const* d_in, const int* in_sizes, int n_in,
                              void* d_out, int out_size, void* d_ws, size_t ws_size,
                              hipStream_t stream) {
    (void)in_sizes; (void)n_in; (void)out_size; (void)ws_size;

    const float* x = (const float*)d_in[0];
    const float* w_ih[4]; const float* w_hh[4];
    const float* b_ih[4]; const float* b_hh[4];
    for (int l = 0; l < 4; ++l) {
        w_ih[l] = (const float*)d_in[1 + 4 * l];
        w_hh[l] = (const float*)d_in[2 + 4 * l];
        b_ih[l] = (const float*)d_in[3 + 4 * l];
        b_hh[l] = (const float*)d_in[4 + 4 * l];
    }
    const float* fc1_w = (const float*)d_in[17];
    const float* fc1_b = (const float*)d_in[18];
    const float* fc2_w = (const float*)d_in[19];
    const float* fc2_b = (const float*)d_in[20];
    float* out = (float*)d_out;

    // ---- workspace carve-up (all chunks are multiples of 256B) ----
    const size_t SZ_BUF = (size_t)BB * TT * HH * sizeof(_Float16);   // 64 MiB
    char* p = (char*)d_ws;
    _Float16* buf0 = (_Float16*)p;  p += SZ_BUF;
    _Float16* buf1 = (_Float16*)p;  p += SZ_BUF;
    float* bias_all = (float*)p;    p += 4 * GG * sizeof(float);
    const int din[4] = {INP, HH, HH, HH};
    uint32_t* wih_pk[4];
    uint32_t* whh_pk[4];
    for (int l = 0; l < 4; ++l) {
        wih_pk[l] = (uint32_t*)p;   p += (size_t)(GG / 16) * (din[l] / 32) * 32 * 8 * 4;
    }
    for (int l = 0; l < 4; ++l) {
        whh_pk[l] = (uint32_t*)p;   p += (size_t)(GG / 16) * (HH / 32) * 32 * 8 * 4;
    }

    // 1) x -> f16
    {
        int n = BB * TT * INP;
        k_f32_to_f16<<<(n + 255) / 256, 256, 0, stream>>>(x, buf0, n);
    }
    // 2) combined biases + packed weights (deterministic, every call)
    for (int l = 0; l < 4; ++l) {
        k_bias_combine<<<(GG + 255) / 256, 256, 0, stream>>>(b_ih[l], b_hh[l],
                                                             bias_all + l * GG);
        int tot_ih = (GG / 16) * (din[l] / 32) * 32 * 8;
        k_pack_B<<<(tot_ih + 255) / 256, 256, 0, stream>>>(w_ih[l], wih_pk[l], din[l]);
        int tot_hh = (GG / 16) * (HH / 32) * 32 * 8;
        k_pack_B<<<(tot_hh + 255) / 256, 256, 0, stream>>>(w_hh[l], whh_pk[l], HH);
    }
    // 3) four persistent LSTM layers (32 WGs x 256 thr each)
    k_lstm_layer<INP><<<BB / 16, 256, 0, stream>>>(buf0, wih_pk[0], whh_pk[0],
                                                   bias_all + 0 * GG, buf1);
    k_lstm_layer<HH> <<<BB / 16, 256, 0, stream>>>(buf1, wih_pk[1], whh_pk[1],
                                                   bias_all + 1 * GG, buf0);
    k_lstm_layer<HH> <<<BB / 16, 256, 0, stream>>>(buf0, wih_pk[2], whh_pk[2],
                                                   bias_all + 2 * GG, buf1);
    k_lstm_layer<HH> <<<BB / 16, 256, 0, stream>>>(buf1, wih_pk[3], whh_pk[3],
                                                   bias_all + 3 * GG, buf0);
    // 4) FC head on h_{T-1}
    k_fc_head<<<(BB + 127) / 128, 128, 0, stream>>>(buf0, fc1_w, fc1_b, fc2_w, fc2_b, out);
}
